// MANN_10737418240134
// MI455X (gfx1250) — matrix operand order repaired and checked by reading, not compiled, not run on hardware
//
#include <hip/hip_runtime.h>

typedef __attribute__((ext_vector_type(16))) __bf16 v16bf;
typedef __attribute__((ext_vector_type(8)))  float  v8f;
typedef __attribute__((ext_vector_type(8)))  unsigned int v8u;

#define BD 256      // input dim D
#define HD 128      // hidden
#define GHD 64      // gating hidden
#define KE 8        // experts
#define MROWS 128   // rows per block (8 waves x 16)
#define NTHREADS 256

// LDS regions (bytes)
#define R0 0        // 32 KB: gw1 / e1w,e2w ping-pong (2x16KB) / x1,x2 A-bounce
#define R1 32768    // 16 KB: h A-bounce / e1b / e2b / head-W ping-pong (2x4KB)
#define R2 49152    // 4 KB : blend [128][8] f32
#define R3 53248    // 2 KB : gw2 B tiles, later head bias
#define LDS_BYTES 55296

// ---- CDNA5 async global->LDS staging (ASYNCcnt) with safe fallback ----
#if defined(__HIP_DEVICE_COMPILE__) && __has_builtin(__builtin_amdgcn_global_load_async_to_lds_b128)
#define ASYNC_OK 1
#else
#define ASYNC_OK 0
#endif

#if ASYNC_OK
typedef int v4i_g __attribute__((vector_size(16)));                 // matches builtin pointee
typedef __attribute__((address_space(1))) v4i_g* as1_v4i_p;         // global (prints as __device__)
typedef __attribute__((address_space(3))) v4i_g* as3_v4i_p;         // LDS
#endif

__device__ __forceinline__ void cp16_async(void* ldsDst, const void* gSrc) {
#if ASYNC_OK
  __builtin_amdgcn_global_load_async_to_lds_b128(
      (as1_v4i_p)const_cast<void*>(gSrc),
      (as3_v4i_p)ldsDst, 0, 0);
#else
  *(uint4*)ldsDst = *(const uint4*)gSrc;
#endif
}
#if ASYNC_OK
# if __has_builtin(__builtin_amdgcn_s_wait_asynccnt)
#  define WAIT_ASYNC(n) __builtin_amdgcn_s_wait_asynccnt(n)
# else
#  define WAIT_ASYNC(n) asm volatile("s_wait_asynccnt %0" :: "n"(n) : "memory")
# endif
#else
# define WAIT_ASYNC(n) ((void)0)
#endif

// ---- fragment swizzles (bf16 WMMA 16x16x32, wave32) ----
// B 32x16 tile: lane = n + 16*((k%16)>=8); pairs 0-3: K%8, pairs 4-7: +16
__device__ __forceinline__ unsigned swzB(int kl, int nl) {
  unsigned lane = (unsigned)((nl & 15) | (((kl >> 3) & 1) << 4));
  unsigned pair = (unsigned)(((kl & 7) >> 1) + ((kl >> 4) << 2));
  return lane * 16u + pair * 2u + (unsigned)(kl & 1);
}
// A 16x32 tile: lane = m + 16*((k%16)>=8); pairs 0-3: K%8, pairs 4-7: +16
__device__ __forceinline__ unsigned swzA(int ml, int kl) {
  unsigned lane = (unsigned)ml + (((kl & 15) >= 8) ? 16u : 0u);
  unsigned pair = (unsigned)(((kl & 7) >> 1) + ((kl >> 4) << 2));
  return lane * 16u + pair * 2u + (unsigned)(kl & 1);
}
__device__ __forceinline__ unsigned short f2bfu(float f) {   // f32 -> bf16 (RNE)
  unsigned u = __float_as_uint(f);
  return (unsigned short)((u + 0x7FFFu + ((u >> 16) & 1u)) >> 16);
}
__device__ __forceinline__ v8f wmma_bf16(v16bf a, v16bf b, v8f c) {
  return __builtin_amdgcn_wmma_f32_16x16x32_bf16(false, a, false, b, (short)0, c, false, false);
}
// per-row (per-lane) scale of an A fragment via native V_PK_MUL_BF16.
// sdup = bf16 scale bits duplicated into both 16-bit lanes of a VGPR.
__device__ __forceinline__ v16bf scale16(v16bf a, unsigned sdup) {
  union { v16bf b; v8u u; } x; x.b = a;
  v8u r;
#pragma unroll
  for (int i = 0; i < 8; ++i) {
    unsigned t;
    asm("v_pk_mul_bf16 %0, %1, %2" : "=v"(t) : "v"(x.u[i]), "v"(sdup));
    r[i] = t;
  }
  union { v8u u; v16bf b; } y; y.u = r;
  return y.b;
}
__device__ __forceinline__ unsigned dupbf(float f) {
  return (unsigned)f2bfu(f) * 0x10001u;
}

// =============== weight pre-pass: f32 -> bf16, B-fragment-swizzled ===============
__global__ void prep_B_normal(const float* __restrict__ src, unsigned short* __restrict__ dst,
                              int Kt, int Nt, int Ksrc, int Nsrc) {
  int total = Kt * 32 * Nt * 16, Nfull = Nt * 16;
  for (int i = blockIdx.x * blockDim.x + threadIdx.x; i < total; i += gridDim.x * blockDim.x) {
    int n = i % Nfull, k = i / Nfull;
    float v = (k < Ksrc && n < Nsrc) ? src[(size_t)k * Nsrc + n] : 0.f;
    dst[(size_t)((k >> 5) * Nt + (n >> 4)) * 512 + swzB(k & 31, n & 15)] = f2bfu(v);
  }
}
__global__ void prep_B_trans(const float* __restrict__ src, unsigned short* __restrict__ dst,
                             int Kt, int Nt, int K, int N, int nExp) {
  int per = Kt * 32 * Nt * 16, Nfull = Nt * 16;
  int total = per * nExp;
  for (int i = blockIdx.x * blockDim.x + threadIdx.x; i < total; i += gridDim.x * blockDim.x) {
    int e = i / per, r = i % per;
    int n = r % Nfull, k = r / Nfull;
    float v = (k < K && n < N) ? src[(size_t)e * K * N + (size_t)n * K + k] : 0.f;
    dst[(size_t)e * per + (size_t)((k >> 5) * Nt + (n >> 4)) * 512 + swzB(k & 31, n & 15)] = f2bfu(v);
  }
}
__global__ void prep_head(const float* __restrict__ pw, const float* __restrict__ vw,
                          unsigned short* __restrict__ dst) {
  int total = 8 * 128 * 16;
  for (int i = blockIdx.x * blockDim.x + threadIdx.x; i < total; i += gridDim.x * blockDim.x) {
    int e = i / (128 * 16), r = i % (128 * 16);
    int n = r % 16, k = r / 16;
    float v = 0.f;
    if (n == 0) v = vw[e * 128 + k];
    else if (n < 4) v = pw[(size_t)e * 3 * 128 + (n - 1) * 128 + k];
    dst[(size_t)e * 2048 + (size_t)(k >> 5) * 512 + swzB(k & 31, n)] = f2bfu(v);
  }
}
__global__ void prep_headbias(const float* __restrict__ pb, const float* __restrict__ vb,
                              unsigned short* __restrict__ dst) {
  int i = blockIdx.x * blockDim.x + threadIdx.x;
  if (i < 512) {
    int n = i % 16, k = i / 16;
    float v = 0.f;
    if (k < 8) { if (n == 0) v = vb[k]; else if (n < 4) v = pb[k * 3 + (n - 1)]; }
    dst[swzB(k, n)] = f2bfu(v);
  }
}

// =============== fused MANN forward ===============
__global__ __launch_bounds__(NTHREADS, 1)
void mann_fused(const float* __restrict__ state,
                const float* __restrict__ gb1, const float* __restrict__ gb2,
                const unsigned short* __restrict__ wsGw1,
                const unsigned short* __restrict__ wsGw2,
                const unsigned short* __restrict__ wsE1w,
                const unsigned short* __restrict__ wsE1b,
                const unsigned short* __restrict__ wsE2w,
                const unsigned short* __restrict__ wsE2b,
                const unsigned short* __restrict__ wsHw,
                const unsigned short* __restrict__ wsHb,
                float* __restrict__ outV, float* __restrict__ outP,
                float* __restrict__ outBlend) {
  __shared__ __align__(16) unsigned char smem[LDS_BYTES];
  const int tid  = threadIdx.x;
  const int lane = tid & 31;
  const int wv   = tid >> 5;        // wave id 0..7 -> owns rows [wv*16, wv*16+16)
  const int mloc = lane & 15;       // M (A layout) / N (B,C layout)
  const int hfl  = lane >> 4;       // lane half
  const int rowBase = blockIdx.x * MROWS;
  const int myRow   = rowBase + wv * 16 + mloc;

  // ---- async-stage gating weights while we convert the state tile
  {
    const uint4* s = (const uint4*)wsGw1;
    for (int i = tid; i < 2048; i += NTHREADS) cp16_async(smem + R0 + (size_t)i * 16, s + i);
    const uint4* s2 = (const uint4*)wsGw2;
    for (int i = tid; i < 128; i += NTHREADS) cp16_async(smem + R3 + (size_t)i * 16, s2 + i);
  }

  // ---- state A fragments (16x32 bf16 x 8), loaded straight into A layout
  v16bf Ast[8];
  {
    const float* rp = state + (size_t)myRow * BD + hfl * 8;
#pragma unroll
    for (int kt = 0; kt < 8; ++kt) {
#pragma unroll
      for (int p = 0; p < 8; ++p) {
        int off = kt * 32 + ((p >> 2) << 4) + ((p & 3) << 1);
        float2 v = *(const float2*)(rp + off);
        Ast[kt][2 * p]     = (__bf16)v.x;
        Ast[kt][2 * p + 1] = (__bf16)v.y;
      }
    }
  }
  WAIT_ASYNC(0);
  __syncthreads();

  // ---- gating layer 1: h = relu(state@gw1 + gb1)
  v8f accg[4] = {};
#pragma unroll
  for (int kt = 0; kt < 8; ++kt)
#pragma unroll
    for (int nt = 0; nt < 4; ++nt) {
      v16bf bfr = *(const v16bf*)(smem + R0 + (size_t)(kt * 4 + nt) * 1024 + lane * 32);
      accg[nt] = wmma_bf16(Ast[kt], bfr, accg[nt]);
    }
  {   // bias+relu, bounce C->A layout through wave-private LDS (R1)
    unsigned short* hdst = (unsigned short*)(smem + R1);
#pragma unroll
    for (int nt = 0; nt < 4; ++nt) {
      float g1 = gb1[nt * 16 + mloc];
#pragma unroll
      for (int r = 0; r < 8; ++r) {
        float v = accg[nt][r] + g1; v = v > 0.f ? v : 0.f;
        int m = hfl * 8 + r, k = nt * 16 + mloc;
        hdst[(unsigned)(wv * 2 + (k >> 5)) * 512u + swzA(m, k & 31)] = f2bfu(v);
      }
    }
  }
  v16bf hA[2];
#pragma unroll
  for (int kt = 0; kt < 2; ++kt)
    hA[kt] = *(const v16bf*)(smem + R1 + (size_t)(wv * 2 + kt) * 1024 + lane * 32);

  // ---- gating layer 2 + softmax -> blend in R2
  v8f accL = {};
#pragma unroll
  for (int kt = 0; kt < 2; ++kt) {
    v16bf bfr = *(const v16bf*)(smem + R3 + (size_t)kt * 1024 + lane * 32);
    accL = wmma_bf16(hA[kt], bfr, accL);
  }
  {
    float* bl = (float*)(smem + R2);
    if (mloc < 8) {
      float g2 = gb2[mloc];
#pragma unroll
      for (int r = 0; r < 8; ++r) bl[(wv * 16 + hfl * 8 + r) * 8 + mloc] = accL[r] + g2;
    }
  }
  __syncthreads();
  if (tid < MROWS) {
    float* bl = (float*)(smem + R2) + tid * 8;
    float mx = bl[0];
#pragma unroll
    for (int e = 1; e < 8; ++e) mx = fmaxf(mx, bl[e]);
    float s = 0.f, ex[8];
#pragma unroll
    for (int e = 0; e < 8; ++e) { ex[e] = __expf(bl[e] - mx); s += ex[e]; }
    float inv = 1.f / s;
    float* ob = outBlend + (size_t)(rowBase + tid) * 8;
#pragma unroll
    for (int e = 0; e < 8; ++e) { float b = ex[e] * inv; bl[e] = b; ob[e] = b; }
  }
  __syncthreads();

  // blend as an A fragment (K = expert index, 8 valid) for bias WMMAs
  v16bf blA = {};
  if (hfl == 0) {
    const float* bl = (const float*)(smem + R2) + (wv * 16 + mloc) * 8;
#pragma unroll
    for (int e = 0; e < 8; ++e) blA[e] = (__bf16)bl[e];
  }
  const float* blendRow = (const float*)(smem + R2) + (wv * 16 + mloc) * 8;

  // stage e1 bias B (8 KB) into R1 (plain copy; consumed after layer-1 loop)
  {
    const uint4* s = (const uint4*)wsE1b; uint4* d = (uint4*)(smem + R1);
    for (int i = tid; i < 512; i += NTHREADS) d[i] = s[i];
  }

  // ---- layer 1: acc1 = sum_e (blend_e o state) @ e1w[e]^T
  // async ping-pong pipeline: 32 stages of 16KB (stage s: expert s>>2, quarter s&3)
  auto issueL1 = [&](int s, int buf) {
    int e = s >> 2, q = s & 3;
    const uint4* se = (const uint4*)wsE1w + (size_t)e * 4096;
    unsigned char* db = smem + R0 + buf * 16384;
    for (int i = tid; i < 1024; i += NTHREADS) {
      int t = i >> 6, w = i & 63;          // dst tile [kt][ntl], src tile [kt][q*2+ntl]
      int kt = t >> 1, ntl = t & 1;
      cp16_async(db + (size_t)i * 16, se + (kt * 8 + q * 2 + ntl) * 64 + w);
    }
  };
  v8f acc1[8] = {};
  v16bf asS[8];
  issueL1(0, 0);
  for (int s = 0; s < 32; ++s) {
    int cur = s & 1;
    if (s + 1 < 32) { issueL1(s + 1, cur ^ 1); WAIT_ASYNC(4); }
    else            { WAIT_ASYNC(0); }
    __syncthreads();
    int e = s >> 2, q = s & 3;
    if (q == 0) {
      unsigned sd = dupbf(blendRow[e]);
#pragma unroll
      for (int kt = 0; kt < 8; ++kt) asS[kt] = scale16(Ast[kt], sd);
    }
    const unsigned char* db = smem + R0 + cur * 16384;
#pragma unroll
    for (int kt = 0; kt < 8; ++kt)
#pragma unroll
      for (int ntl = 0; ntl < 2; ++ntl) {
        v16bf bfr = *(const v16bf*)(db + (size_t)(kt * 2 + ntl) * 1024 + lane * 32);
        acc1[q * 2 + ntl] = wmma_bf16(asS[kt], bfr, acc1[q * 2 + ntl]);
      }
    __syncthreads();
  }
  // blended bias (rank-8 WMMA) + relu
#pragma unroll
  for (int nt = 0; nt < 8; ++nt) {
    v16bf bfr = *(const v16bf*)(smem + R1 + (size_t)nt * 1024 + lane * 32);
    acc1[nt] = wmma_bf16(blA, bfr, acc1[nt]);
#pragma unroll
    for (int r = 0; r < 8; ++r) acc1[nt][r] = acc1[nt][r] > 0.f ? acc1[nt][r] : 0.f;
  }
  // x1 C->A bounce through R0 (wave-private slice)
  {
    unsigned short* xdst = (unsigned short*)(smem + R0);
#pragma unroll
    for (int nt = 0; nt < 8; ++nt) {
      int k = nt * 16 + mloc;
#pragma unroll
      for (int r = 0; r < 8; ++r)
        xdst[(unsigned)(wv * 4 + (k >> 5)) * 512u + swzA(hfl * 8 + r, k & 31)] = f2bfu(acc1[nt][r]);
    }
  }
  v16bf A2[4];
#pragma unroll
  for (int kt = 0; kt < 4; ++kt)
    A2[kt] = *(const v16bf*)(smem + R0 + (size_t)(wv * 4 + kt) * 1024 + lane * 32);
  __syncthreads();
  {   // stage e2 bias into R1
    const uint4* s = (const uint4*)wsE2b; uint4* d = (uint4*)(smem + R1);
    for (int i = tid; i < 512; i += NTHREADS) d[i] = s[i];
  }

  // ---- layer 2: 16 stages of 16KB (stage s: expert s>>1, half s&1)
  auto issueL2 = [&](int s, int buf) {
    int e = s >> 1, h = s & 1;
    const uint4* se = (const uint4*)wsE2w + (size_t)e * 2048;
    unsigned char* db = smem + R0 + buf * 16384;
    for (int i = tid; i < 1024; i += NTHREADS) {
      int t = i >> 6, w = i & 63;          // dst tile [kt][ntl], src tile [kt][h*4+ntl]
      int kt = t >> 2, ntl = t & 3;
      cp16_async(db + (size_t)i * 16, se + (kt * 8 + h * 4 + ntl) * 64 + w);
    }
  };
  v8f acc2[8] = {};
  v16bf asS2[4];
  issueL2(0, 0);
  for (int s = 0; s < 16; ++s) {
    int cur = s & 1;
    if (s + 1 < 16) { issueL2(s + 1, cur ^ 1); WAIT_ASYNC(4); }
    else            { WAIT_ASYNC(0); }
    __syncthreads();
    int e = s >> 1, h = s & 1;
    if (h == 0) {
      unsigned sd = dupbf(blendRow[e]);
#pragma unroll
      for (int kt = 0; kt < 4; ++kt) asS2[kt] = scale16(A2[kt], sd);
    }
    const unsigned char* db = smem + R0 + cur * 16384;
#pragma unroll
    for (int kt = 0; kt < 4; ++kt)
#pragma unroll
      for (int ntl = 0; ntl < 4; ++ntl) {
        v16bf bfr = *(const v16bf*)(db + (size_t)(kt * 4 + ntl) * 1024 + lane * 32);
        acc2[h * 4 + ntl] = wmma_bf16(asS2[kt], bfr, acc2[h * 4 + ntl]);
      }
    __syncthreads();
  }
#pragma unroll
  for (int nt = 0; nt < 8; ++nt) {
    v16bf bfr = *(const v16bf*)(smem + R1 + (size_t)nt * 1024 + lane * 32);
    acc2[nt] = wmma_bf16(blA, bfr, acc2[nt]);
#pragma unroll
    for (int r = 0; r < 8; ++r) acc2[nt][r] = acc2[nt][r] > 0.f ? acc2[nt][r] : 0.f;
  }
  // x2 C->A bounce through R0 (wave-private slice)
  {
    unsigned short* xdst = (unsigned short*)(smem + R0);
#pragma unroll
    for (int nt = 0; nt < 8; ++nt) {
      int k = nt * 16 + mloc;
#pragma unroll
      for (int r = 0; r < 8; ++r)
        xdst[(unsigned)(wv * 4 + (k >> 5)) * 512u + swzA(hfl * 8 + r, k & 31)] = f2bfu(acc2[nt][r]);
    }
  }
  v16bf A3[4];
#pragma unroll
  for (int kt = 0; kt < 4; ++kt)
    A3[kt] = *(const v16bf*)(smem + R0 + (size_t)(wv * 4 + kt) * 1024 + lane * 32);
  __syncthreads();
  {   // head bias (1 KB) -> R3 (gw2 is dead), plain copy
    const uint4* s = (const uint4*)wsHb; uint4* d = (uint4*)(smem + R3);
    for (int i = tid; i < 64; i += NTHREADS) d[i] = s[i];
  }

  // ---- heads: [value | policy0..2] in one N tile; 8 stages of 4KB, ping-pong in R1
  auto issueH = [&](int e, int buf) {
    const uint4* se = (const uint4*)wsHw + (size_t)e * 256;
    unsigned char* db = smem + R1 + buf * 4096;
    for (int i = tid; i < 256; i += NTHREADS) cp16_async(db + (size_t)i * 16, se + i);
  };
  v8f acch = {};
  issueH(0, 0);
  for (int e = 0; e < KE; ++e) {
    int cur = e & 1;
    if (e + 1 < KE) { issueH(e + 1, cur ^ 1); WAIT_ASYNC(1); }
    else            { WAIT_ASYNC(0); }
    __syncthreads();
    unsigned sd = dupbf(blendRow[e]);
    const unsigned char* db = smem + R1 + cur * 4096;
#pragma unroll
    for (int kt = 0; kt < 4; ++kt) {
      v16bf as = scale16(A3[kt], sd);
      v16bf bfr = *(const v16bf*)(db + (size_t)kt * 1024 + lane * 32);
      acch = wmma_bf16(as, bfr, acch);
    }
    __syncthreads();
  }
  {
    v16bf bfr = *(const v16bf*)(smem + R3 + lane * 32);
    acch = wmma_bf16(blA, bfr, acch);
  }
  if (mloc < 4) {
#pragma unroll
    for (int r = 0; r < 8; ++r) {
      int R = rowBase + wv * 16 + hfl * 8 + r;
      float v = acch[r];
      if (mloc == 0) outV[R] = v;
      else outP[(size_t)R * 3 + (mloc - 1)] = v;
    }
  }
}

extern "C" void kernel_launch(void* const* d_in, const int* in_sizes, int n_in,
                              void* d_out, int out_size, void* d_ws, size_t ws_size,
                              hipStream_t stream) {
  const float* state = (const float*)d_in[0];
  const float* gw1 = (const float*)d_in[1];
  const float* gb1 = (const float*)d_in[2];
  const float* gw2 = (const float*)d_in[3];
  const float* gb2 = (const float*)d_in[4];
  const float* e1w = (const float*)d_in[5];
  const float* e1b = (const float*)d_in[6];
  const float* e2w = (const float*)d_in[7];
  const float* e2b = (const float*)d_in[8];
  const float* pw  = (const float*)d_in[9];
  const float* pb  = (const float*)d_in[10];
  const float* vw  = (const float*)d_in[11];
  const float* vb  = (const float*)d_in[12];
  const int Bv = in_sizes[0] / BD;

  // carve d_ws (bf16, pre-swizzled weights): ~871 KB total
  unsigned short* ws    = (unsigned short*)d_ws;
  unsigned short* wsGw1 = ws;               // 8*4*512
  unsigned short* wsGw2 = wsGw1 + 16384;    // 2*512
  unsigned short* wsE1b = wsGw2 + 1024;     // 8*512
  unsigned short* wsE2b = wsE1b + 4096;     // 8*512
  unsigned short* wsHb  = wsE2b + 4096;     // 512
  unsigned short* wsHw  = wsHb  + 512;      // 8*4*512
  unsigned short* wsE1w = wsHw  + 16384;    // 8*8*8*512
  unsigned short* wsE2w = wsE1w + 262144;   // 8*4*8*512

  prep_B_normal<<<64,  256, 0, stream>>>(gw1, wsGw1, 8, 4, BD, GHD);
  prep_B_normal<<<4,   256, 0, stream>>>(gw2, wsGw2, 2, 1, GHD, KE);
  prep_B_normal<<<16,  256, 0, stream>>>(e1b, wsE1b, 1, 8, KE, HD);
  prep_B_normal<<<16,  256, 0, stream>>>(e2b, wsE2b, 1, 8, KE, HD);
  prep_B_trans <<<512, 256, 0, stream>>>(e1w, wsE1w, 8, 8, BD, HD, KE);
  prep_B_trans <<<256, 256, 0, stream>>>(e2w, wsE2w, 4, 8, HD, HD, KE);
  prep_head    <<<64,  256, 0, stream>>>(pw, vw, wsHw);
  prep_headbias<<<2,   256, 0, stream>>>(pb, vb, wsHb);

  float* out = (float*)d_out;
  float* outV = out;                       // value  [B]
  float* outP = out + Bv;                  // policy [B,3]
  float* outBlend = out + (size_t)4 * Bv;  // blend  [B,8]
  mann_fused<<<Bv / MROWS, NTHREADS, 0, stream>>>(
      state, gb1, gb2, wsGw1, wsGw2, wsE1w, wsE1b, wsE2w, wsE2b, wsHw, wsHb,
      outV, outP, outBlend);
  (void)n_in; (void)out_size; (void)ws_size;
}
